// Net_27118423507319
// MI455X (gfx1250) — matrix operand order repaired and checked by reading, not compiled, not run on hardware
//
#include <hip/hip_runtime.h>
#include <cmath>

// ---------------------------------------------------------------------------
// CDNA5 (gfx1250) wave32 WMMA types
// ---------------------------------------------------------------------------
typedef __attribute__((ext_vector_type(16))) _Float16 v16h;
typedef __attribute__((ext_vector_type(8)))  _Float16 v8h;
typedef __attribute__((ext_vector_type(4)))  _Float16 v4h;
typedef __attribute__((ext_vector_type(8)))  float    v8f;
typedef __attribute__((ext_vector_type(4)))  float    v4f;

// ---------------------------------------------------------------------------
// Fragment loaders (16-bit WMMA layouts per CDNA5 ISA §7.12.2).
//   A 16x32 f16 : lane%16 = M; elements 0..7 -> K = kb + 8*hi + e,
//                 elements 8..15 -> K = kb + 16 + 8*hi + (e-8)   (hi = lane>=16)
//   => per lane two contiguous 8-half runs -> 2x ds_load_b128.
//   B 32x16 f16 : lane%16 = N; same K mapping as A (8*hi + e, 16 + 8*hi + (e-8))
//   => with weights stored transposed [col][k] in LDS, two contiguous
//      8-half runs -> 2x ds_load_b128.
//   C/D 16x16 f32: lane%16 = N, vgpr r -> M = r + 8*hi.
// ---------------------------------------------------------------------------
__device__ __forceinline__ v16h load_A_frag(const _Float16* tile, int kbase, int ld) {
  const int lane = threadIdx.x & 31;
  const _Float16* base = tile + (lane & 15) * ld + kbase + ((lane >> 4) << 3);
  union { v16h v; v8h h[2]; } u;
  u.h[0] = *(const v8h*)(base);
  u.h[1] = *(const v8h*)(base + 16);
  return u.v;
}

__device__ __forceinline__ v16h load_B_frag_t(const _Float16* wT, int kbase, int nbase, int ld) {
  const int lane = threadIdx.x & 31;
  const _Float16* base = wT + (nbase + (lane & 15)) * ld + kbase + ((lane >> 4) << 3);
  union { v16h v; v8h h[2]; } u;
  u.h[0] = *(const v8h*)(base);
  u.h[1] = *(const v8h*)(base + 16);
  return u.v;
}

__device__ __forceinline__ float elu_f(float x) { return x > 0.f ? x : expm1f(x); }

// ---------------------------------------------------------------------------
// Edge / pool scatter: each wave owns one edge; 32 lanes cover 64 features
// (2 per lane). Atomic f32 adds land in the L2-resident agg buffer
// (100k x 64 x 4B = 25.6 MB << 192 MB L2). Index streams are prefetched
// ahead (global_prefetch_b8).
// ---------------------------------------------------------------------------
__global__ __launch_bounds__(256) void edge_scatter(
    const float* __restrict__ h, const int* __restrict__ src,
    const int* __restrict__ dst, float* __restrict__ agg, int nEdges)
{
  const int t = blockIdx.x * 256 + threadIdx.x;
  const int e = t >> 5;           // wave-uniform edge id
  if (e >= nEdges) return;
  const int f = t & 31;
  if (f == 0) {
    __builtin_prefetch(src + e + 1024, 0, 0);
    __builtin_prefetch(dst + e + 1024, 0, 0);
  }
  const int s = src[e];
  const int d = dst[e];
  const float* hr = h + (size_t)s * 64;
  float*       ar = agg + (size_t)d * 64;
  atomicAdd(ar + f,      hr[f]);
  atomicAdd(ar + f + 32, hr[f + 32]);
}

// ---------------------------------------------------------------------------
// Fused GraphConv:  hout = ELU( [hin | agg] (Nx128) @ [Wroot;Wrel] (128x64) + b )
// Block = 256 threads = 8 waves; wave owns a 16x64 output tile.
// K=128 -> 4 WMMA K-steps x 4 col tiles = 16 v_wmma_f32_16x16x32_f16 per wave.
// ---------------------------------------------------------------------------
#define LDK 136   // transposed weights: K stride (128 + 8 pad), rows 272B (16B-aligned)
#define LDA 136   // A tile: K stride (128 + 8 pad)

__global__ __launch_bounds__(256) void graphconv_wmma(
    const float* __restrict__ hin, const float* __restrict__ agg,
    const float* __restrict__ wroot, const float* __restrict__ wrel,
    const float* __restrict__ bias, float* __restrict__ hout, int n)
{
  __shared__ __attribute__((aligned(16))) _Float16 sWT[64 * LDK];     // 17 KB [col][k]
  __shared__ __attribute__((aligned(16))) _Float16 sA[8][16 * LDA];   // 34 KB per-wave 16x128

  // Stage stacked [Wroot; Wrel] transposed to [col][k] as f16, 4 k per store.
  for (int g = threadIdx.x; g < 64 * 32; g += 256) {
    const int c  = g & 63;
    const int k0 = (g >> 6) << 2;       // 0..124 step 4
    v4h pk;
#pragma unroll
    for (int j = 0; j < 4; ++j) {
      const int k = k0 + j;
      const float w = (k < 64) ? wroot[k * 64 + c] : wrel[(k - 64) * 64 + c];
      pk[j] = (_Float16)w;
    }
    *(v4h*)(&sWT[c * LDK + k0]) = pk;
  }
  __syncthreads();

  const int wave = threadIdx.x >> 5;
  const int lane = threadIdx.x & 31;
  const int row0 = (blockIdx.x * 8 + wave) * 16;

  // Stage per-wave A tile: 16 rows x [hin(64) | agg(64)] as f16, 4 halfs/store.
  {
    const int r    = lane >> 1;
    const int half = lane & 1;          // 0 -> hin cols, 1 -> agg cols
    const int grow = row0 + r;
    _Float16* dstp = &sA[wave][r * LDA + half * 64];
    if (grow < n) {
      const float* srcp = (half ? agg : hin) + (size_t)grow * 64;
#pragma unroll
      for (int c = 0; c < 64; c += 4) {
        const float4 v = *(const float4*)(srcp + c);
        v4h pk;
        pk[0] = (_Float16)v.x; pk[1] = (_Float16)v.y;
        pk[2] = (_Float16)v.z; pk[3] = (_Float16)v.w;
        *(v4h*)(dstp + c) = pk;
      }
    } else {
      const v4h z = {};
#pragma unroll
      for (int c = 0; c < 64; c += 4) *(v4h*)(dstp + c) = z;
    }
  }
  // same-wave LDS ordering (DScnt) makes the tile visible without a barrier

  v8f acc[4] = {};
#pragma unroll
  for (int s = 0; s < 4; ++s) {
    const v16h a = load_A_frag(sA[wave], 32 * s, LDA);
#pragma unroll
    for (int t = 0; t < 4; ++t) {
      const v16h b = load_B_frag_t(sWT, 32 * s, 16 * t, LDK);
      acc[t] = __builtin_amdgcn_wmma_f32_16x16x32_f16(
          false, a, false, b, (short)0, acc[t], false, false);
    }
  }

  // Bias + ELU + store (C layout: lane%16 = col, vgpr r -> row r + 8*hi)
#pragma unroll
  for (int t = 0; t < 4; ++t) {
    const int col = t * 16 + (lane & 15);
    const float bv = bias[col];
#pragma unroll
    for (int r = 0; r < 8; ++r) {
      const int row = row0 + r + ((lane >> 4) << 3);
      if (row < n) hout[(size_t)row * 64 + col] = elu_f(acc[t][r] + bv);
    }
  }
}

// ---------------------------------------------------------------------------
// MLP head: per wave 16 set-rows.
//   fc1 64->64 (8 WMMA) + ELU, fc2 64->32 (4 WMMA) + ELU,
//   fc3 32->2 + log_softmax on VALU from an LDS tile.
// ---------------------------------------------------------------------------
#define LDK1 72   // transposed fc1/fc2 weights: K stride (64 + 8 pad), 144B rows
#define LDA2 72   // pooled/h1 A tile: K stride
#define LDH  36   // fc2 output tile: 144B rows (16B-aligned for v4f loads)

__global__ __launch_bounds__(256) void mlp_head_wmma(
    const float* __restrict__ pooled,
    const float* __restrict__ fc1w, const float* __restrict__ fc1b,
    const float* __restrict__ fc2w, const float* __restrict__ fc2b,
    const float* __restrict__ fc3w, const float* __restrict__ fc3b,
    float* __restrict__ out, int S)
{
  __shared__ __attribute__((aligned(16))) _Float16 sW1T[64 * LDK1];   // 9 KB [col][k]
  __shared__ __attribute__((aligned(16))) _Float16 sW2T[32 * LDK1];   // 4.5 KB [col][k]
  __shared__ __attribute__((aligned(16))) _Float16 sA[8][16 * LDA2];  // 18 KB
  __shared__ __attribute__((aligned(16))) float    sH[8][16 * LDH];   // 18 KB

  for (int g = threadIdx.x; g < 64 * 16; g += 256) {   // W1: 64 cols x 16 k-quads
    const int c  = g & 63;
    const int k0 = (g >> 6) << 2;
    v4h pk;
#pragma unroll
    for (int j = 0; j < 4; ++j) pk[j] = (_Float16)fc1w[(k0 + j) * 64 + c];
    *(v4h*)(&sW1T[c * LDK1 + k0]) = pk;
  }
  for (int g = threadIdx.x; g < 32 * 16; g += 256) {   // W2: 32 cols x 16 k-quads
    const int c  = g & 31;
    const int k0 = (g >> 5) << 2;
    v4h pk;
#pragma unroll
    for (int j = 0; j < 4; ++j) pk[j] = (_Float16)fc2w[(k0 + j) * 32 + c];
    *(v4h*)(&sW2T[c * LDK1 + k0]) = pk;
  }
  __syncthreads();

  const int wave = threadIdx.x >> 5;
  const int lane = threadIdx.x & 31;
  const int row0 = (blockIdx.x * 8 + wave) * 16;

  // Stage pooled tile 16x64 as f16
  {
    const int r    = lane >> 1;
    const int half = lane & 1;
    const int grow = row0 + r;
    _Float16* dstp = &sA[wave][r * LDA2 + half * 32];
    if (grow < S) {
      const float* srcp = pooled + (size_t)grow * 64 + half * 32;
#pragma unroll
      for (int c = 0; c < 32; c += 4) {
        const float4 v = *(const float4*)(srcp + c);
        v4h pk;
        pk[0] = (_Float16)v.x; pk[1] = (_Float16)v.y;
        pk[2] = (_Float16)v.z; pk[3] = (_Float16)v.w;
        *(v4h*)(dstp + c) = pk;
      }
    } else {
      const v4h z = {};
#pragma unroll
      for (int c = 0; c < 32; c += 4) *(v4h*)(dstp + c) = z;
    }
  }

  // fc1: 16x64 = A(16x64) @ W1(64x64)
  v8f acc1[4] = {};
#pragma unroll
  for (int s = 0; s < 2; ++s) {
    const v16h a = load_A_frag(sA[wave], 32 * s, LDA2);
#pragma unroll
    for (int t = 0; t < 4; ++t) {
      const v16h b = load_B_frag_t(sW1T, 32 * s, 16 * t, LDK1);
      acc1[t] = __builtin_amdgcn_wmma_f32_16x16x32_f16(
          false, a, false, b, (short)0, acc1[t], false, false);
    }
  }
  // ELU; redistribute C-layout -> A-layout through the same per-wave LDS tile
#pragma unroll
  for (int t = 0; t < 4; ++t) {
    const int col = t * 16 + (lane & 15);
    const float bv = fc1b[col];
#pragma unroll
    for (int r = 0; r < 8; ++r) {
      const int m = r + ((lane >> 4) << 3);
      sA[wave][m * LDA2 + col] = (_Float16)elu_f(acc1[t][r] + bv);
    }
  }

  // fc2: 16x32 = A(16x64) @ W2(64x32)
  v8f acc2[2] = {};
#pragma unroll
  for (int s = 0; s < 2; ++s) {
    const v16h a = load_A_frag(sA[wave], 32 * s, LDA2);
#pragma unroll
    for (int t = 0; t < 2; ++t) {
      const v16h b = load_B_frag_t(sW2T, 32 * s, 16 * t, LDK1);
      acc2[t] = __builtin_amdgcn_wmma_f32_16x16x32_f16(
          false, a, false, b, (short)0, acc2[t], false, false);
    }
  }
#pragma unroll
  for (int t = 0; t < 2; ++t) {
    const int col = t * 16 + (lane & 15);
    const float bv = fc2b[col];
#pragma unroll
    for (int r = 0; r < 8; ++r) {
      const int m = r + ((lane >> 4) << 3);
      sH[wave][m * LDH + col] = elu_f(acc2[t][r] + bv);
    }
  }

  // fc3 (32->2) + log_softmax; lanes 0..15 each own one set-row
  if (lane < 16) {
    const int row = row0 + lane;
    if (row < S) {
      float z0 = fc3b[0], z1 = fc3b[1];
      const v4f* hp = (const v4f*)(&sH[wave][lane * LDH]);
#pragma unroll
      for (int q = 0; q < 8; ++q) {
        const v4f h4 = hp[q];
#pragma unroll
        for (int j = 0; j < 4; ++j) {
          const int k = q * 4 + j;
          const float2 w2 = *(const float2*)(fc3w + k * 2);
          z0 += h4[j] * w2.x;
          z1 += h4[j] * w2.y;
        }
      }
      const float mx  = fmaxf(z0, z1);
      const float lse = mx + logf(expf(z0 - mx) + expf(z1 - mx));
      out[(size_t)row * 2 + 0] = z0 - lse;
      out[(size_t)row * 2 + 1] = z1 - lse;
    }
  }
}

// ---------------------------------------------------------------------------
// Host-side launcher (graph-capture safe: only async ops on `stream`)
// ---------------------------------------------------------------------------
extern "C" void kernel_launch(void* const* d_in, const int* in_sizes, int n_in,
                              void* d_out, int out_size, void* d_ws, size_t ws_size,
                              hipStream_t stream)
{
  const float* x          = (const float*)d_in[0];
  const int*   edge_src   = (const int*)d_in[1];
  const int*   edge_dst   = (const int*)d_in[2];
  const int*   gather_idx = (const int*)d_in[3];
  const int*   seg_idx    = (const int*)d_in[4];
  const float* w_root0    = (const float*)d_in[5];
  const float* w_rel0     = (const float*)d_in[6];
  const float* b0         = (const float*)d_in[7];
  const float* w_root_h   = (const float*)d_in[8];
  const float* w_rel_h    = (const float*)d_in[9];
  const float* b_h        = (const float*)d_in[10];
  const float* fc1_w      = (const float*)d_in[11];
  const float* fc1_b      = (const float*)d_in[12];
  const float* fc2_w      = (const float*)d_in[13];
  const float* fc2_b      = (const float*)d_in[14];
  const float* fc3_w      = (const float*)d_in[15];
  const float* fc3_b      = (const float*)d_in[16];

  const int N = in_sizes[0] / 64;
  const int E = in_sizes[1];
  const int A = in_sizes[3];
  const int S = out_size / 2;

  float* agg    = (float*)d_ws;                  // [N,64]
  float* hA     = agg + (size_t)N * 64;          // [N,64]
  float* hB     = hA  + (size_t)N * 64;          // [N,64]
  float* pooled = agg;                           // [S,64] reuses agg after conv3

  const int convBlocks = (N + 127) / 128;
  const int scatBlocks = (E * 32 + 255) / 256;
  const int poolBlocks = (A * 32 + 255) / 256;
  const int mlpBlocks  = (S + 127) / 128;

  // ---- GraphConv layer 0 ----
  hipMemsetAsync(agg, 0, (size_t)N * 64 * sizeof(float), stream);
  edge_scatter<<<scatBlocks, 256, 0, stream>>>(x, edge_src, edge_dst, agg, E);
  graphconv_wmma<<<convBlocks, 256, 0, stream>>>(x, agg, w_root0, w_rel0, b0, hA, N);

  // ---- GraphConv layer 1 ----
  hipMemsetAsync(agg, 0, (size_t)N * 64 * sizeof(float), stream);
  edge_scatter<<<scatBlocks, 256, 0, stream>>>(hA, edge_src, edge_dst, agg, E);
  graphconv_wmma<<<convBlocks, 256, 0, stream>>>(hA, agg, w_root_h, w_rel_h, b_h, hB, N);

  // ---- GraphConv layer 2 ----
  hipMemsetAsync(agg, 0, (size_t)N * 64 * sizeof(float), stream);
  edge_scatter<<<scatBlocks, 256, 0, stream>>>(hB, edge_src, edge_dst, agg, E);
  graphconv_wmma<<<convBlocks, 256, 0, stream>>>(hB, agg,
      w_root_h + 64 * 64, w_rel_h + 64 * 64, b_h + 64, hA, N);

  // ---- Set pooling ----
  hipMemsetAsync(pooled, 0, (size_t)S * 64 * sizeof(float), stream);
  edge_scatter<<<poolBlocks, 256, 0, stream>>>(hA, gather_idx, seg_idx, pooled, A);

  // ---- MLP head ----
  mlp_head_wmma<<<mlpBlocks, 256, 0, stream>>>(pooled,
      fc1_w, fc1_b, fc2_w, fc2_b, fc3_w, fc3_b, (float*)d_out, S);
}
